// MambaLM_81295140978780
// MI455X (gfx1250) — matrix operand order repaired and checked by reading, not compiled, not run on hardware
//
#include <hip/hip_runtime.h>
#include <math.h>

#define V_SIZE   32000
#define E_DIM    1024
#define NLAYER   4
#define NSTATE   16
#define DCONV    4
#define DINNER   2048
#define DTRANK   64
#define BSZn     2
#define SEQ      512
#define ROWS     (BSZn*SEQ)            /* 1024 */
#define XPROJ_N  (DTRANK + 2*NSTATE)   /* 96  */

#ifndef USE_ASYNC_LDS
#define USE_ASYNC_LDS 1
#endif

typedef __bf16 bf16_t;
typedef __attribute__((ext_vector_type(16))) bf16_t v16bf;
typedef __attribute__((ext_vector_type(8)))  float  v8f;

static __device__ __forceinline__ unsigned short f32_to_bf16(float f) {
  unsigned int u = __float_as_uint(f);
  unsigned int r = u + 0x7FFFu + ((u >> 16) & 1u);   // round-to-nearest-even
  return (unsigned short)(r >> 16);
}

// ---- gfx1250 async global->LDS copy (ASYNCcnt path), with plain-LDS fallback
static __device__ __forceinline__ void async_g2l_b128(const void* gptr, void* lptr) {
#if USE_ASYNC_LDS
  unsigned lds_off = (unsigned)(unsigned long long)(uintptr_t)lptr;  // low 32b = LDS byte offset
  unsigned long long ga = (unsigned long long)(uintptr_t)gptr;
  asm volatile("global_load_async_to_lds_b128 %0, %1, off"
               :: "v"(lds_off), "v"(ga) : "memory");
#else
  *reinterpret_cast<uint4*>(lptr) = *reinterpret_cast<const uint4*>(gptr);
#endif
}
static __device__ __forceinline__ void async_wait_le4() {
#if USE_ASYNC_LDS
  asm volatile("s_wait_asynccnt 0x4" ::: "memory");
#endif
}
static __device__ __forceinline__ void async_wait_le0() {
#if USE_ASYNC_LDS
  asm volatile("s_wait_asynccnt 0x0" ::: "memory");
#endif
}

// ---------------------------------------------------------------- converters
__global__ void k_f32_to_bf16(const float* __restrict__ src,
                              unsigned short* __restrict__ dst, long long n) {
  long long i = (long long)blockIdx.x * blockDim.x + threadIdx.x;
  if (i < n) dst[i] = f32_to_bf16(src[i]);
}

// ---------------------------------------------------------------- embedding
__global__ void k_embed(const int* __restrict__ ids, const float* __restrict__ emb,
                        float* __restrict__ x) {
  int i = blockIdx.x * blockDim.x + threadIdx.x;
  if (i >= ROWS * E_DIM) return;
  int row = i / E_DIM, c = i % E_DIM;
  x[i] = emb[(size_t)ids[row] * E_DIM + c];
}

// ---------------------------------------------------------------- layernorm -> bf16
__global__ void k_layernorm_bf16(const float* __restrict__ x,
                                 const float* __restrict__ w,
                                 const float* __restrict__ b,
                                 unsigned short* __restrict__ out) {
  int row = blockIdx.x;
  int tid = threadIdx.x;
  const float* xr = x + (size_t)row * E_DIM;
  __shared__ float red[256];
  float s = 0.f;
  for (int i = tid; i < E_DIM; i += 256) s += xr[i];
  red[tid] = s; __syncthreads();
  for (int off = 128; off > 0; off >>= 1) { if (tid < off) red[tid] += red[tid+off]; __syncthreads(); }
  float mu = red[0] / (float)E_DIM;
  __syncthreads();
  float v = 0.f;
  for (int i = tid; i < E_DIM; i += 256) { float d = xr[i] - mu; v += d*d; }
  red[tid] = v; __syncthreads();
  for (int off = 128; off > 0; off >>= 1) { if (tid < off) red[tid] += red[tid+off]; __syncthreads(); }
  float inv = rsqrtf(red[0] / (float)E_DIM + 1e-5f);
  for (int i = tid; i < E_DIM; i += 256)
    out[(size_t)row * E_DIM + i] = f32_to_bf16((xr[i] - mu) * inv * w[i] + b[i]);
}

// ---------------------------------------------------------------- WMMA bf16 GEMM
// C[M,N] = A[M,K] * B[N,K]^T.  Block = 256 thr = 8 waves, macro-tile 128x128.
// Wave grid 4(M) x 2(N): each wave owns 32x64 = 8 accumulators.
// Double-buffered LDS; K staged in 32-slices with async global->LDS DMA
// overlapped against the WMMAs of the previous slice. Last slice is peeled so
// the steady-state loop is branch-free.
// Caller guarantees B has at least ceil(N/128)*128 readable rows (pad in ws).
__global__ void __launch_bounds__(256)
k_gemm_bf16_wmma(const unsigned short* __restrict__ A,
                 const unsigned short* __restrict__ B,
                 float* __restrict__ C,
                 int M, int N, int K, int lda, int ldb, int ldc) {
  __shared__ unsigned short sA[2][128 * 32];   // 2 x 8 KB
  __shared__ unsigned short sB[2][128 * 32];   // 2 x 8 KB
  int tid   = threadIdx.x;
  int lane  = tid & 31;
  int w     = tid >> 5;
  int half  = lane >> 4;
  int l16   = lane & 15;
  int blockM = blockIdx.y * 128;
  int blockN = blockIdx.x * 128;
  int wm = (w >> 1) * 32;                   // 0,32,64,96
  int wn = (w & 1) * 64;                    // 0,64

  // this thread's two 16B staging chunks: chunk c -> row c>>2, col (c&3)*8
  int c0 = tid, c1 = tid + 256;
  int r0 = c0 >> 2, q0 = (c0 & 3) * 8;
  int r1 = c1 >> 2, q1 = (c1 & 3) * 8;
  const unsigned short* gA0 = A + (size_t)(blockM + r0) * lda + q0;
  const unsigned short* gA1 = A + (size_t)(blockM + r1) * lda + q1;
  const unsigned short* gB0 = B + (size_t)(blockN + r0) * ldb + q0;
  const unsigned short* gB1 = B + (size_t)(blockN + r1) * ldb + q1;

  auto stage = [&](int buf, int k0) {
    async_g2l_b128(gA0 + k0, &sA[buf][r0 * 32 + q0]);
    async_g2l_b128(gA1 + k0, &sA[buf][r1 * 32 + q1]);
    async_g2l_b128(gB0 + k0, &sB[buf][r0 * 32 + q0]);
    async_g2l_b128(gB1 + k0, &sB[buf][r1 * 32 + q1]);
  };

  v8f acc[2][4];
  #pragma unroll
  for (int i = 0; i < 2; ++i)
    #pragma unroll
    for (int j = 0; j < 4; ++j)
      acc[i][j] = v8f{0.f,0.f,0.f,0.f,0.f,0.f,0.f,0.f};

  auto compute = [&](int cur) {
    union Frag { uint4 q[2]; v16bf v; };
    Frag fa[2], fb[4];
    const unsigned short* bA = sA[cur];
    const unsigned short* bB = sB[cur];
    #pragma unroll
    for (int i = 0; i < 2; ++i) {
      const unsigned short* pa = bA + (wm + i * 16 + l16) * 32;
      fa[i].q[0] = *reinterpret_cast<const uint4*>(pa + half * 8);        // K=8h+j
      fa[i].q[1] = *reinterpret_cast<const uint4*>(pa + 16 + half * 8);   // K=16+8h+j
    }
    #pragma unroll
    for (int j = 0; j < 4; ++j) {
      const unsigned short* pb = bB + (wn + j * 16 + l16) * 32;
      fb[j].q[0] = *reinterpret_cast<const uint4*>(pb + half * 8);        // K=8h+j
      fb[j].q[1] = *reinterpret_cast<const uint4*>(pb + 16 + half * 8);
    }
    #pragma unroll
    for (int i = 0; i < 2; ++i)
      #pragma unroll
      for (int j = 0; j < 4; ++j)
        acc[i][j] = __builtin_amdgcn_wmma_f32_16x16x32_bf16(
            false, fa[i].v, false, fb[j].v, (short)0, acc[i][j], false, false);
  };

  int nk = K >> 5;                          // K is a multiple of 32; nk >= 2 here
  stage(0, 0);
  // steady state: unconditional stage(next) -> wait<=4 -> compute(cur)
  for (int ks = 0; ks < nk - 1; ++ks) {
    int cur = ks & 1;
    stage(cur ^ 1, (ks + 1) << 5);          // DMA next slice under this slice's math
    async_wait_le4();                       // slice ks landed; ks+1 in flight
    __syncthreads();
    compute(cur);
    __syncthreads();                        // reads done before buffer re-staged
  }
  // peeled last slice
  async_wait_le0();
  __syncthreads();
  compute((nk - 1) & 1);

  // C/D layout: VGPR r -> M = r + 8*half, N = lane%16
  #pragma unroll
  for (int i = 0; i < 2; ++i)
    #pragma unroll
    for (int j = 0; j < 4; ++j) {
      int ncol = blockN + wn + j * 16 + l16;
      if (ncol >= N) continue;
      #pragma unroll
      for (int r = 0; r < 8; ++r) {
        int mrow = blockM + wm + i * 16 + r + half * 8;
        C[(size_t)mrow * ldc + ncol] = acc[i][j][r];
      }
    }
}

// ---------------------------------------------------------------- causal conv (DC=4) + SiLU
__global__ void k_conv_silu(const float* __restrict__ xz, const float* __restrict__ cw,
                            const float* __restrict__ cb, float* __restrict__ u,
                            unsigned short* __restrict__ ub) {
  int i = blockIdx.x * blockDim.x + threadIdx.x;
  if (i >= ROWS * DINNER) return;
  int d   = i % DINNER;
  int row = i / DINNER;
  int l   = row % SEQ;
  int b   = row / SEQ;
  float acc = cb[d];
  #pragma unroll
  for (int k = 0; k < DCONV; ++k) {
    int lt = l + k - (DCONV - 1);
    if (lt >= 0)
      acc += cw[d * DCONV + k] * xz[((size_t)(b * SEQ + lt)) * (2 * DINNER) + d];
  }
  float s = acc / (1.f + __expf(-acc));
  u[i] = s;
  ub[i] = f32_to_bf16(s);
}

// ---------------------------------------------------------------- softplus(x + bias)
__global__ void k_softplus_bias(float* __restrict__ dl, const float* __restrict__ dtb) {
  int i = blockIdx.x * blockDim.x + threadIdx.x;
  if (i >= ROWS * DINNER) return;
  float x = dl[i] + dtb[i % DINNER];
  dl[i] = (x > 20.f) ? x : log1pf(__expf(x));
}

// ---------------------------------------------------------------- selective scan
__global__ void k_scan(const float* __restrict__ delta, const float* __restrict__ u,
                       const float* __restrict__ xdbl, const float* __restrict__ a_log,
                       float* __restrict__ y) {
  int tid = threadIdx.x;
  int b   = blockIdx.x / (DINNER / 256);
  int d   = (blockIdx.x % (DINNER / 256)) * 256 + tid;
  float a[NSTATE], h[NSTATE];
  #pragma unroll
  for (int n = 0; n < NSTATE; ++n) { a[n] = -__expf(a_log[(size_t)d * NSTATE + n]); h[n] = 0.f; }
  __shared__ float sB[NSTATE], sC[NSTATE];
  for (int t = 0; t < SEQ; ++t) {
    int row = b * SEQ + t;
    if (tid < NSTATE)            sB[tid]          = xdbl[(size_t)row * XPROJ_N + DTRANK + tid];
    else if (tid < 2 * NSTATE)   sC[tid - NSTATE] = xdbl[(size_t)row * XPROJ_N + DTRANK + NSTATE + (tid - NSTATE)];
    __syncthreads();
    float dt = delta[(size_t)row * DINNER + d];
    float uv = u[(size_t)row * DINNER + d];
    float acc = 0.f;
    #pragma unroll
    for (int n = 0; n < NSTATE; ++n) {
      h[n] = __expf(dt * a[n]) * h[n] + dt * sB[n] * uv;
      acc += h[n] * sC[n];
    }
    y[(size_t)row * DINNER + d] = acc;
    __syncthreads();
  }
}

// ---------------------------------------------------------------- gating: (y + u*D) * silu(res)
__global__ void k_gate(const float* __restrict__ yscan, const float* __restrict__ u,
                       const float* __restrict__ xz, const float* __restrict__ Dp,
                       unsigned short* __restrict__ yb) {
  int i = blockIdx.x * blockDim.x + threadIdx.x;
  if (i >= ROWS * DINNER) return;
  int d = i % DINNER, row = i / DINNER;
  float res = xz[(size_t)row * (2 * DINNER) + DINNER + d];
  float val = (yscan[i] + u[i] * Dp[d]) * (res / (1.f + __expf(-res)));
  yb[i] = f32_to_bf16(val);
}

// ---------------------------------------------------------------- residual add
__global__ void k_add(float* __restrict__ x, const float* __restrict__ h, int n) {
  int i = blockIdx.x * blockDim.x + threadIdx.x;
  if (i < n) x[i] += h[i];
}

// ---------------------------------------------------------------- loss
__global__ void k_zero(float* p) { if (threadIdx.x == 0 && blockIdx.x == 0) *p = 0.f; }

__global__ void k_loss(const float* __restrict__ logits, const int* __restrict__ tgt,
                       float* __restrict__ acc) {
  int row = blockIdx.x;
  int tid = threadIdx.x;
  const float* lr = logits + (size_t)row * V_SIZE;
  __shared__ float red[256];
  float m = -3.4e38f;
  for (int i = tid; i < V_SIZE; i += 256) m = fmaxf(m, lr[i]);
  red[tid] = m; __syncthreads();
  for (int off = 128; off > 0; off >>= 1) { if (tid < off) red[tid] = fmaxf(red[tid], red[tid+off]); __syncthreads(); }
  float mx = red[0];
  __syncthreads();
  float s = 0.f;
  for (int i = tid; i < V_SIZE; i += 256) s += __expf(lr[i] - mx);
  red[tid] = s; __syncthreads();
  for (int off = 128; off > 0; off >>= 1) { if (tid < off) red[tid] += red[tid+off]; __syncthreads(); }
  if (tid == 0) {
    int t = tgt[row];
    float nll = 0.f;
    if (t >= 0) nll = -(lr[t] - mx - logf(red[0]));
    atomicAdd(acc, nll);
  }
}

__global__ void k_loss_final(const float* __restrict__ acc, float* __restrict__ out) {
  if (threadIdx.x == 0 && blockIdx.x == 0) out[0] = acc[0] / (float)ROWS;
}

// ================================================================ launch
extern "C" void kernel_launch(void* const* d_in, const int* in_sizes, int n_in,
                              void* d_out, int out_size, void* d_ws, size_t ws_size,
                              hipStream_t stream) {
  (void)in_sizes; (void)n_in; (void)out_size; (void)ws_size;
  const int*   ids   = (const int*)d_in[0];
  const int*   tgt   = (const int*)d_in[1];
  const float* emb   = (const float*)d_in[2];
  const float* ln_w  = (const float*)d_in[3];
  const float* ln_b  = (const float*)d_in[4];
  const float* w_in  = (const float*)d_in[5];
  const float* cw    = (const float*)d_in[6];
  const float* cb    = (const float*)d_in[7];
  const float* w_x   = (const float*)d_in[8];
  const float* w_dt  = (const float*)d_in[9];
  const float* dtb   = (const float*)d_in[10];
  const float* a_log = (const float*)d_in[11];
  const float* Dp    = (const float*)d_in[12];
  const float* w_out = (const float*)d_in[13];
  const float* flnw  = (const float*)d_in[14];
  const float* flnb  = (const float*)d_in[15];

  char* ws = (char*)d_ws;
  size_t off = 0;
  auto alloc = [&](size_t bytes) -> char* {
    char* p = ws + off; off += (bytes + 255) & ~(size_t)255; return p;
  };
  unsigned short* emb_bf   = (unsigned short*)alloc((size_t)V_SIZE * E_DIM * 2);
  unsigned short* win_bf   = (unsigned short*)alloc((size_t)NLAYER * 2 * DINNER * E_DIM * 2);
  // +32 rows of pad so the branch-free GEMM may read (never store) past N=96
  unsigned short* wx_bf    = (unsigned short*)alloc(((size_t)NLAYER * XPROJ_N + 32) * DINNER * 2);
  unsigned short* wdt_bf   = (unsigned short*)alloc((size_t)NLAYER * DINNER * DTRANK * 2);
  unsigned short* wout_bf  = (unsigned short*)alloc((size_t)NLAYER * E_DIM * DINNER * 2);
  float*          xres     = (float*)alloc((size_t)ROWS * E_DIM * 4);
  unsigned short* hnorm_bf = (unsigned short*)alloc((size_t)ROWS * E_DIM * 2);
  float*          xz       = (float*)alloc((size_t)ROWS * 2 * DINNER * 4);
  float*          u        = (float*)alloc((size_t)ROWS * DINNER * 4);
  unsigned short* u_bf     = (unsigned short*)alloc((size_t)ROWS * DINNER * 2);
  float*          xdbl     = (float*)alloc((size_t)ROWS * XPROJ_N * 4);
  unsigned short* xdbl_bf  = (unsigned short*)alloc((size_t)ROWS * XPROJ_N * 2);
  float*          delta    = (float*)alloc((size_t)ROWS * DINNER * 4);
  float*          yscan    = (float*)alloc((size_t)ROWS * DINNER * 4);
  unsigned short* y_bf     = (unsigned short*)alloc((size_t)ROWS * DINNER * 2);
  float*          hout     = (float*)alloc((size_t)ROWS * E_DIM * 4);
  unsigned short* xfin_bf  = (unsigned short*)alloc((size_t)ROWS * E_DIM * 2);
  float*          lacc     = (float*)alloc(256);

  float* logits = (float*)d_out;
  float* loss   = logits + (size_t)ROWS * V_SIZE;

  dim3 blk(256);
  auto cvt = [&](const float* s, unsigned short* d, long long n) {
    k_f32_to_bf16<<<dim3((unsigned)((n + 255) / 256)), blk, 0, stream>>>(s, d, n);
  };
  cvt(emb,   emb_bf,  (long long)V_SIZE * E_DIM);
  cvt(w_in,  win_bf,  (long long)NLAYER * 2 * DINNER * E_DIM);
  cvt(w_x,   wx_bf,   (long long)NLAYER * XPROJ_N * DINNER);
  cvt(w_dt,  wdt_bf,  (long long)NLAYER * DINNER * DTRANK);
  cvt(w_out, wout_bf, (long long)NLAYER * E_DIM * DINNER);

  k_embed<<<dim3((ROWS * E_DIM + 255) / 256), blk, 0, stream>>>(ids, emb, xres);

  auto gemm = [&](const unsigned short* A, const unsigned short* B, float* C,
                  int M, int N, int K, int lda, int ldb, int ldc) {
    dim3 grid((unsigned)((N + 127) / 128), (unsigned)(M / 128));
    k_gemm_bf16_wmma<<<grid, blk, 0, stream>>>(A, B, C, M, N, K, lda, ldb, ldc);
  };

  for (int l = 0; l < NLAYER; ++l) {
    k_layernorm_bf16<<<dim3(ROWS), blk, 0, stream>>>(xres, ln_w + (size_t)l * E_DIM,
                                                     ln_b + (size_t)l * E_DIM, hnorm_bf);
    // in_proj: (1024,1024) x (4096,1024)^T -> (1024,4096)
    gemm(hnorm_bf, win_bf + (size_t)l * 2 * DINNER * E_DIM, xz,
         ROWS, 2 * DINNER, E_DIM, E_DIM, E_DIM, 2 * DINNER);
    k_conv_silu<<<dim3((ROWS * DINNER + 255) / 256), blk, 0, stream>>>(
        xz, cw + (size_t)l * DINNER * DCONV, cb + (size_t)l * DINNER, u, u_bf);
    // x_proj: (1024,2048) x (96,2048)^T -> (1024,96)  (B padded to 128 rows)
    gemm(u_bf, wx_bf + (size_t)l * XPROJ_N * DINNER, xdbl,
         ROWS, XPROJ_N, DINNER, DINNER, DINNER, XPROJ_N);
    cvt(xdbl, xdbl_bf, (long long)ROWS * XPROJ_N);
    // dt_proj: (1024,64 of 96-stride) x (2048,64)^T -> (1024,2048)
    gemm(xdbl_bf, wdt_bf + (size_t)l * DINNER * DTRANK, delta,
         ROWS, DINNER, DTRANK, XPROJ_N, DTRANK, DINNER);
    k_softplus_bias<<<dim3((ROWS * DINNER + 255) / 256), blk, 0, stream>>>(
        delta, dtb + (size_t)l * DINNER);
    k_scan<<<dim3(BSZn * (DINNER / 256)), blk, 0, stream>>>(
        delta, u, xdbl, a_log + (size_t)l * DINNER * NSTATE, yscan);
    k_gate<<<dim3((ROWS * DINNER + 255) / 256), blk, 0, stream>>>(
        yscan, u, xz, Dp + (size_t)l * DINNER, y_bf);
    // out_proj: (1024,2048) x (1024,2048)^T -> (1024,1024)
    gemm(y_bf, wout_bf + (size_t)l * E_DIM * DINNER, hout,
         ROWS, E_DIM, DINNER, DINNER, DINNER, E_DIM);
    k_add<<<dim3((ROWS * E_DIM + 255) / 256), blk, 0, stream>>>(xres, hout, ROWS * E_DIM);
  }

  k_layernorm_bf16<<<dim3(ROWS), blk, 0, stream>>>(xres, flnw, flnb, xfin_bf);
  // logits: (1024,1024) x (32000,1024)^T -> (1024,32000)   (bf16 emb fits in 192MB L2)
  gemm(xfin_bf, emb_bf, logits, ROWS, V_SIZE, E_DIM, E_DIM, E_DIM, V_SIZE);

  k_zero<<<dim3(1), blk, 0, stream>>>(lacc);
  k_loss<<<dim3(ROWS), blk, 0, stream>>>(logits, tgt, lacc);
  k_loss_final<<<dim3(1), blk, 0, stream>>>(lacc, loss);
}
